// SchNetLayer_58480274702583
// MI455X (gfx1250) — matrix operand order repaired and checked by reading, not compiled, not run on hardware
//
#include <hip/hip_runtime.h>
#include <hip/hip_bf16.h>

typedef __attribute__((ext_vector_type(16))) _Float16 v16h;
typedef __attribute__((ext_vector_type(8)))  float    v8f;

#define N_ATOMS 1024
#define NF      128
#define N_RBF   20

union AFrag { v16h h; _Float16 e[16]; unsigned u[8]; };
union PK2   { _Float16 f[2]; unsigned u; };

__device__ __forceinline__ float fast_tanh(float x) {
#if __has_builtin(__builtin_amdgcn_tanhf)
  return __builtin_amdgcn_tanhf(x);      // hardware V_TANH_F32 (gfx1250)
#else
  float e = __expf(2.f * x);
  return __builtin_fmaf(-2.f, __builtin_amdgcn_rcpf(e + 1.f), 1.f);
#endif
}

// ---------------------------------------------------------------------------
// Prep: swizzle W1 (20x128, K padded to 32) and W2 (128x128) into WMMA
// B-matrix fragment order (per-lane packed u32 = 2 f16), and compute
// Sfeat[f] = sum_j feat[j][f] (for the b2 bias term of the aggregation).
// B 32x16 layout: lanes 0-15 hold K=0..7,16..23 (VGPR v<4 -> K=2v,2v+1;
//                 v>=4 -> K=8+2v,9+2v), lanes 16-31 the same +8.
// ---------------------------------------------------------------------------
__global__ void schnet_prep(const float* __restrict__ W1,
                            const float* __restrict__ W2,
                            const float* __restrict__ feat,
                            unsigned* __restrict__ w1s,   // 8 tiles * 256 u32
                            unsigned* __restrict__ w2s,   // 32 tiles * 256 u32
                            float* __restrict__ Sfeat) {
  int t = blockIdx.x * blockDim.x + threadIdx.x;
  if (t < 2048) {                 // W1 fragments: nt in 0..7
    int nt = t >> 8, rem = t & 255, v = rem >> 5, lane = rem & 31;
    int s = lane >> 4, n = lane & 15;
    int K0 = 2 * v + 8 * s + (v >= 4 ? 8 : 0);
    int col = nt * 16 + n;
    PK2 p;
    p.f[0] = (K0     < N_RBF) ? (_Float16)W1[K0 * NF + col]       : (_Float16)0.f;
    p.f[1] = (K0 + 1 < N_RBF) ? (_Float16)W1[(K0 + 1) * NF + col] : (_Float16)0.f;
    w1s[t] = p.u;
  } else if (t < 2048 + 8192) {   // W2 fragments: (nt,ks) tiles
    int q = t - 2048;
    int tile = q >> 8, rem = q & 255, v = rem >> 5, lane = rem & 31;
    int nt = tile >> 2, ks = tile & 3;
    int s = lane >> 4, n = lane & 15;
    int K0 = 32 * ks + 2 * v + 8 * s + (v >= 4 ? 8 : 0);
    int col = nt * 16 + n;
    PK2 p;
    p.f[0] = (_Float16)W2[K0 * NF + col];
    p.f[1] = (_Float16)W2[(K0 + 1) * NF + col];
    w2s[q] = p.u;
  } else if (t < 2048 + 8192 + NF) {  // Sfeat
    int f = t - (2048 + 8192);
    float s = 0.f;
    for (int j = 0; j < N_ATOMS; ++j) s += feat[j * NF + f];
    Sfeat[f] = s;
  }
}

// ---------------------------------------------------------------------------
// Fused main: one workgroup per atom i, 8 waves; wave w owns feature tile
// nt=w. Loop over j in chunks of 16 pairs:
//   dist -> RBF (A frag in regs, quadratic form: 2 FMA + exp per slot)
//   -> WMMA vs W1 -> +b1, hw tanh -> LDS f16 (double buffered, 1 barrier)
//   -> 4x WMMA vs W2 (K=128) -> filters * feat[j] -> accumulate agg[i,:].
// ---------------------------------------------------------------------------
__global__ __launch_bounds__(256) void schnet_main(
    const float* __restrict__ feat, const float* __restrict__ pos,
    const float* __restrict__ centers, const float* __restrict__ widths,
    const float* __restrict__ b1,
    const unsigned* __restrict__ w1s, const unsigned* __restrict__ w2s,
    float* __restrict__ agg) {
  __shared__ __attribute__((aligned(16))) _Float16 hbuf[2][16 * NF];  // 2x4 KB

  const int i    = blockIdx.x;
  const int lane = threadIdx.x & 31;
  const int nt   = threadIdx.x >> 5;   // feature tile (wave id)
  const int s    = lane >> 4;          // half-wave (K sub-block)
  const int p    = lane & 15;          // pair row within tile
  const int col  = nt * 16 + p;        // output feature column

  // Persistent B fragments (coalesced u32 loads from pre-swizzled buffers)
  AFrag w1f;
#pragma unroll
  for (int v = 0; v < 8; ++v) w1f.u[v] = w1s[nt * 256 + v * 32 + lane];
  AFrag w2f[4];
#pragma unroll
  for (int ks = 0; ks < 4; ++ks)
#pragma unroll
    for (int v = 0; v < 8; ++v)
      w2f[ks].u[v] = w2s[(nt * 4 + ks) * 256 + v * 32 + lane];

  // Hoisted RBF params in quadratic form:
  //   exp(-((d-c)/w)^2) = exp(aq*d2 + bq*d + cq),
  //   aq = -1/w^2, bq = 2c/w^2, cq = -c^2/w^2.
  // Slot A covers K=8s+t (always < N_RBF); slot B covers K=16+8s+t.
  float aqA[8], bqA[8], cqA[8], aqB[8], bqB[8], cqB[8];
#pragma unroll
  for (int t = 0; t < 8; ++t) {
    {
      int r = 8 * s + t;
      float c = centers[r];
      float iw = __builtin_amdgcn_rcpf(widths[r]);
      float iw2 = iw * iw;
      aqA[t] = -iw2; bqA[t] = 2.f * c * iw2; cqA[t] = -c * c * iw2;
    }
    int r2 = 16 + 8 * s + t;
    if (r2 < N_RBF) {
      float c = centers[r2];
      float iw = __builtin_amdgcn_rcpf(widths[r2]);
      float iw2 = iw * iw;
      aqB[t] = -iw2; bqB[t] = 2.f * c * iw2; cqB[t] = -c * c * iw2;
    } else {
      aqB[t] = 0.f; bqB[t] = 0.f; cqB[t] = 0.f;
    }
  }

  const float b1c = b1[col];
  const float pix = pos[i * 3 + 0], piy = pos[i * 3 + 1], piz = pos[i * 3 + 2];

  float acc = 0.f;

  for (int jc = 0; jc < N_ATOMS / 16; ++jc) {
    const int jbase = jc * 16;
    _Float16* hb = hbuf[jc & 1];

    // distance for pair row p (both half-waves compute the same d)
    const int jp = jbase + p;
    float dx = pix - pos[jp * 3 + 0];
    float dy = piy - pos[jp * 3 + 1];
    float dz = piz - pos[jp * 3 + 2];
    float d2 = dx * dx + dy * dy + dz * dz;
    float d  = (d2 > 0.f) ? __builtin_amdgcn_sqrtf(d2) : 0.f;

    // RBF A-fragment: exp(aq*d2 + bq*d + cq); invalid K slots -> 0
    AFrag a;
#pragma unroll
    for (int t = 0; t < 8; ++t) {
      float z1 = __builtin_fmaf(aqA[t], d2, __builtin_fmaf(bqA[t], d, cqA[t]));
      a.e[t] = (_Float16)__expf(z1);
      float z2 = __builtin_fmaf(aqB[t], d2, __builtin_fmaf(bqB[t], d, cqB[t]));
      float rv2 = __expf(z2);
      a.e[8 + t] = (16 + 8 * s + t < N_RBF) ? (_Float16)rv2 : (_Float16)0.f;
    }

    // h = tanh(rbf @ W1 + b1)  -> LDS as f16 [pair][g]
    v8f zc = {};
    v8f hc = __builtin_amdgcn_wmma_f32_16x16x32_f16(
        false, a.h, false, w1f.h, (short)0, zc, false, false);
#pragma unroll
    for (int v = 0; v < 8; ++v) {
      float th = fast_tanh(hc[v] + b1c);
      hb[(v + 8 * s) * NF + col] = (_Float16)th;
    }
    __syncthreads();

    // filters = h @ W2 : reload A-fragments from LDS, chain 4 k-steps
    v8f fc = {};
#pragma unroll
    for (int ks = 0; ks < 4; ++ks) {
      AFrag af;
      *(uint4*)&af.u[0] = *(const uint4*)&hb[p * NF + 32 * ks + 8 * s];
      *(uint4*)&af.u[4] = *(const uint4*)&hb[p * NF + 32 * ks + 16 + 8 * s];
      fc = __builtin_amdgcn_wmma_f32_16x16x32_f16(
          false, af.h, false, w2f[ks].h, (short)0, fc, false, false);
    }

    // aggregated[i,col] += sum_rows filters[row,col] * feat[jbase+row, col]
#pragma unroll
    for (int v = 0; v < 8; ++v) {
      int row = v + 8 * s;
      acc += fc[v] * feat[(jbase + row) * NF + col];
    }
    // next iteration writes the OTHER hbuf half; this chunk's buffer is only
    // reused two chunks later, after the next barrier -> single barrier/chunk
  }

  // fold the two half-wave row groups (rows 0-7 in lanes<16, 8-15 in lanes>=16)
  acc += __shfl_xor(acc, 16, 32);
  if (lane < 16) agg[i * NF + col] = acc;
}

// ---------------------------------------------------------------------------
// Epilogue: out = feat + (agg + b2*Sfeat) @ Wi + bi          (tiny GEMM)
// ---------------------------------------------------------------------------
__global__ void schnet_final(const float* __restrict__ feat,
                             const float* __restrict__ agg,
                             const float* __restrict__ Sfeat,
                             const float* __restrict__ b2,
                             const float* __restrict__ Wi,
                             const float* __restrict__ bi,
                             float* __restrict__ out) {
  __shared__ float arow[NF];
  int i = blockIdx.x, f = threadIdx.x;
  arow[f] = agg[i * NF + f] + b2[f] * Sfeat[f];
  __syncthreads();
  float acc = feat[i * NF + f] + bi[f];
  for (int g = 0; g < NF; ++g) acc += arow[g] * Wi[g * NF + f];
  out[i * NF + f] = acc;
}

// ---------------------------------------------------------------------------
extern "C" void kernel_launch(void* const* d_in, const int* in_sizes, int n_in,
                              void* d_out, int out_size, void* d_ws, size_t ws_size,
                              hipStream_t stream) {
  const float* feat    = (const float*)d_in[0];
  const float* pos     = (const float*)d_in[1];
  const float* centers = (const float*)d_in[2];
  const float* widths  = (const float*)d_in[3];
  const float* W1      = (const float*)d_in[4];
  const float* b1      = (const float*)d_in[5];
  const float* W2      = (const float*)d_in[6];
  const float* b2      = (const float*)d_in[7];
  const float* Wi      = (const float*)d_in[8];
  const float* bi      = (const float*)d_in[9];
  float* out = (float*)d_out;

  char* ws = (char*)d_ws;
  unsigned* w1s  = (unsigned*)(ws);                  //  8 KB
  unsigned* w2s  = (unsigned*)(ws + 8192);           // 32 KB
  float*    Sfe  = (float*)   (ws + 8192 + 32768);   // 512 B
  float*    agg  = (float*)   (ws + 65536);          // 512 KB

  schnet_prep<<<dim3((2048 + 8192 + NF + 255) / 256), dim3(256), 0, stream>>>(
      W1, W2, feat, w1s, w2s, Sfe);
  schnet_main<<<dim3(N_ATOMS), dim3(256), 0, stream>>>(
      feat, pos, centers, widths, b1, w1s, w2s, agg);
  schnet_final<<<dim3(N_ATOMS), dim3(NF), 0, stream>>>(
      feat, agg, Sfe, b2, Wi, bi, out);
}
